// ColorHistogramLoss_52733608460423
// MI455X (gfx1250) — compile-verified, matrix operand order checked by reading, not run on hardware
//
#include <hip/hip_runtime.h>
#include <hip/hip_bf16.h>

typedef __attribute__((ext_vector_type(16))) _Float16 v16h;
typedef __attribute__((ext_vector_type(8)))  float    v8f;

#define BINS        64
#define HW          65536          // 256*256 pixels per (b,c) slice
#define NSLICE      12             // B*C = 4*3
#define CHUNKS      32             // workgroups per slice
#define PIX_PER_WG  (HW / CHUNKS)  // 2048
#define WAVES       8              // 256 threads = 8 wave32
#define PIX_PER_WAVE (PIX_PER_WG / WAVES) // 256
#define TILES       (PIX_PER_WAVE / 32)   // 8 tiles of 32 pixels per wave

// Raw v_exp_f32 (flush-to-zero below 2^-126). The OCML exp2f adds a ~5-op
// denormal fixup per call that is pure overhead here: weights are cast to
// f16 (flushes below ~6e-8) and summed into O(1e3) histograms, so sub-1e-38
// accuracy is meaningless.
#if __has_builtin(__builtin_amdgcn_exp2f)
#define FAST_EXP2(x) __builtin_amdgcn_exp2f(x)
#else
#define FAST_EXP2(x) exp2f(x)
#endif

// ---------------------------------------------------------------- zero ws
__global__ void ch_zero(float* __restrict__ ws, int n) {
    int i = blockIdx.x * blockDim.x + threadIdx.x;
    if (i < n) ws[i] = 0.0f;
}

// ---------------------------------------------------------------- phase 1
// Soft-histogram accumulation. Grid: 2 * NSLICE * CHUNKS blocks of 256.
// Each wave processes 8 tiles of 32 pixels; per tile it builds four
// 32(pixel) x 16(bin) f16 weight tiles (B-matrix layout) and reduces them
// over the pixel (K) dimension with V_WMMA_F32_16X16X32_F16 against an
// all-ones A matrix. f32 partial histograms are flushed with atomics.
__global__ __launch_bounds__(256) void ch_hist(const float* __restrict__ pred,
                                               const float* __restrict__ targ,
                                               float* __restrict__ hist) {
    __shared__ __align__(16) float tile[PIX_PER_WG];

    const int wg     = blockIdx.x;
    const int chunk  = wg % CHUNKS;
    const int slice  = (wg / CHUNKS) % NSLICE;
    const int tensor = wg / (CHUNKS * NSLICE);   // 0 = pred, 1 = target

    const float* __restrict__ src  = tensor ? targ : pred;
    const float* __restrict__ base = src + slice * HW + chunk * PIX_PER_WG;

    // Stage this workgroup's 2048 pixels into LDS (coalesced).
    #pragma unroll
    for (int i = 0; i < PIX_PER_WG / 256; ++i)
        tile[threadIdx.x + i * 256] = base[threadIdx.x + i * 256];
    __syncthreads();

    const int lane = threadIdx.x & 31;
    const int wave = threadIdx.x >> 5;
    const int half = lane >> 4;       // 0: pixels K=0..15, 1: K=16..31
    const int col  = lane & 15;       // bin column N within each 16-bin group

    const float sigma = 1.0f / 64.0f;
    // exp(-(d^2)/(2*sigma^2)) = exp2(d^2 * S2),  S2 = -log2(e)/(2*sigma^2)
    const float S2 = -1.4426950408889634f * 2048.0f;

    float cg[4];
    #pragma unroll
    for (int g = 0; g < 4; ++g)
        cg[g] = ((float)(g * 16 + col) + 0.5f) * sigma;   // bin centers

    v16h ones;
    #pragma unroll
    for (int i = 0; i < 16; ++i) ones[i] = (_Float16)1.0f;

    v8f acc[4] = { {}, {}, {}, {} };   // four 16-bin accumulator tiles (f32)

    #pragma unroll
    for (int t = 0; t < TILES; ++t) {
        // This lane's 16 shared pixel values (LDS broadcast reads, 4x b128).
        const int off = wave * PIX_PER_WAVE + t * 32 + half * 16;
        const float4* p4 = reinterpret_cast<const float4*>(&tile[off]);
        float xv[16];
        #pragma unroll
        for (int q = 0; q < 4; ++q) {
            float4 v = p4[q];
            xv[4 * q + 0] = v.x; xv[4 * q + 1] = v.y;
            xv[4 * q + 2] = v.z; xv[4 * q + 3] = v.w;
        }

        #pragma unroll
        for (int g = 0; g < 4; ++g) {
            v16h b;   // B tile: column N = col, rows K = half*16 + j
            #pragma unroll
            for (int j = 0; j < 16; ++j) {
                float d = xv[j] - cg[g];
                float w = FAST_EXP2(d * d * S2);   // raw v_exp_f32
                b[j] = (_Float16)w;
            }
            // D = ones(16x32) * B(32x16) + C : every row of D is the
            // pixel-sum of B's columns. Accumulates across tiles.
            acc[g] = __builtin_amdgcn_wmma_f32_16x16x32_f16(
                false, ones, false, b, (short)0, acc[g], false, false);
        }
    }

    // Row 0 (lanes 0-15) and row 8 (lanes 16-31) of D live in VGPR 0 of the
    // accumulator; all rows are identical, so each half-wave flushes two of
    // the four 16-bin groups without duplication.
    float* __restrict__ h = hist + (tensor * NSLICE + slice) * BINS;
    if (half == 0) {
        atomicAdd(&h[ 0 + col], acc[0][0]);
        atomicAdd(&h[16 + col], acc[1][0]);
    } else {
        atomicAdd(&h[32 + col], acc[2][0]);
        atomicAdd(&h[48 + col], acc[3][0]);
    }
}

// ---------------------------------------------------------------- phase 2
// Normalize -> CDF (inclusive scan) -> mean |cdf_pred - cdf_target|.
// One block of 64 threads (one thread per bin); 768 values total.
__global__ __launch_bounds__(64) void ch_finalize(const float* __restrict__ hist,
                                                  float* __restrict__ out) {
    __shared__ float sb[BINS];
    const int tid = threadIdx.x;
    float total = 0.0f;

    for (int s = 0; s < NSLICE; ++s) {
        float cdf[2];
        #pragma unroll
        for (int k = 0; k < 2; ++k) {
            float v = hist[(k * NSLICE + s) * BINS + tid];
            sb[tid] = v;
            __syncthreads();
            // Hillis-Steele inclusive scan over 64 bins.
            for (int o = 1; o < BINS; o <<= 1) {
                float add = (tid >= o) ? sb[tid - o] : 0.0f;
                __syncthreads();
                sb[tid] += add;
                __syncthreads();
            }
            float sum = sb[BINS - 1];            // total mass of this hist
            cdf[k] = sb[tid] / (sum + 1e-8f);    // cumsum(h)/(sum) == cumsum(h/sum)
            __syncthreads();
        }
        total += fabsf(cdf[0] - cdf[1]);
    }

    sb[tid] = total;
    __syncthreads();
    for (int o = 32; o > 0; o >>= 1) {
        if (tid < o) sb[tid] += sb[tid + o];
        __syncthreads();
    }
    if (tid == 0) out[0] = sb[0] / (float)(NSLICE * BINS);
}

// ---------------------------------------------------------------- launch
extern "C" void kernel_launch(void* const* d_in, const int* in_sizes, int n_in,
                              void* d_out, int out_size, void* d_ws, size_t ws_size,
                              hipStream_t stream) {
    const float* pred = (const float*)d_in[0];
    const float* targ = (const float*)d_in[1];
    float* hist = (float*)d_ws;        // 2 * 12 * 64 = 1536 floats (6 KB)
    float* out  = (float*)d_out;

    const int histN = 2 * NSLICE * BINS;
    ch_zero<<<(histN + 255) / 256, 256, 0, stream>>>(hist, histN);
    ch_hist<<<2 * NSLICE * CHUNKS, 256, 0, stream>>>(pred, targ, hist);
    ch_finalize<<<1, 64, 0, stream>>>(hist, out);
}
